// MultiHeadAttention_74071005987160
// MI455X (gfx1250) — compile-verified, matrix operand order checked by reading, not run on hardware
//
#include <hip/hip_runtime.h>
#include <hip/hip_bf16.h>

// Problem constants (match reference)
#define BATCH  4
#define SEQ    1024
#define DMODEL 1024
#define NHEAD  16
#define DHEAD  64
#define NREL   2047   // 2*SEQ - 1
#define EKROWS 2048   // Ek padded (+1 zero row): max index l0+jb+lane = 2047
#define EVT_STRIDE 2112 // Evt row stride, zero padded: max index l0+1055 = 2063 < 2112

typedef __bf16 bf16_t;
typedef __attribute__((ext_vector_type(16))) __bf16 v16bf;
typedef __attribute__((ext_vector_type(8)))  float  v8f;

// ---------------------------------------------------------------------------
// WMMA helpers.  CDNA5 v_wmma_f32_16x16x32_bf16: D(16x16 f32) = A(16x32) x B(32x16) + C.
// A layout (ISA 7.12.2): lane m = t&15, hi = t>>4; element i of v16bf holds
//   k = (i<8 ? hi*8 + i : 16 + hi*8 + (i-8))   -> two contiguous 8-elem (16B) chunks.
// B layout mirrors A with n = t&15 (column n of B = one source row when the
// source is stored transposed), same k striping.
// C: element r of v8f -> row (r + 8*hi), col (t&15).
// ---------------------------------------------------------------------------
__device__ inline v8f wmma_bf16(v16bf a, v16bf b, v8f c) {
  return __builtin_amdgcn_wmma_f32_16x16x32_bf16(false, a, false, b, (short)0, c,
                                                 false, false);
}

// Load one 32-wide k-step for this lane from a k-contiguous row (16B aligned).
__device__ inline v16bf load_tile16(const bf16_t* p, int hi) {
  union { uint4 u[2]; v16bf v; } t;
  t.u[0] = *(const uint4*)(p + hi * 8);
  t.u[1] = *(const uint4*)(p + 16 + hi * 8);
  return t.v;
}

// ---------------------------------------------------------------------------
// Prep kernels
// ---------------------------------------------------------------------------
__global__ void fill0_bf16_kernel(bf16_t* __restrict__ p, int n) {
  int i = blockIdx.x * 256 + threadIdx.x;
  if (i < n) p[i] = (bf16_t)0.0f;
}

__global__ void cvt_kernel(const float* __restrict__ src, bf16_t* __restrict__ dst, int n) {
  int i = blockIdx.x * 256 + threadIdx.x;
  if (i < n) dst[i] = (bf16_t)src[i];
}

// dst[c*dstStride + r] = src[r*C + c]
__global__ void transpose_cvt_kernel(const float* __restrict__ src, bf16_t* __restrict__ dst,
                                     int R, int C, int dstStride) {
  int i = blockIdx.x * 256 + threadIdx.x;
  if (i < R * C) {
    int r = i / C, c = i % C;
    dst[(size_t)c * dstStride + r] = (bf16_t)src[i];
  }
}

// ---------------------------------------------------------------------------
// bf16 WMMA GEMM:  C(M=4096, N=1024) = A(M,1024) * Bt(N,1024)^T + bias
// Register-blocked: each wave computes a 16x64 tile (4 accumulators, shared A).
// mode 0: store bf16 head-split  [B,H,L,DH]   (Q, K)
// mode 1: store bf16 head-split transposed [B,H,DH,L]  (V^T)
// mode 2: store f32 row-major [M,N]           (final output projection)
// Block = 128 threads (4 waves) -> 16x256 per block; grid (M/16, N/256).
// ---------------------------------------------------------------------------
__global__ void gemm_bf16_kernel(const bf16_t* __restrict__ A, const bf16_t* __restrict__ Bt,
                                 const float* __restrict__ bias, void* __restrict__ out,
                                 int mode) {
  const int t = threadIdx.x & 31, w = threadIdx.x >> 5;
  const int lane16 = t & 15, hi = t >> 4;
  const int n0 = blockIdx.y * 256 + w * 64;
  const int mrow = blockIdx.x * 16 + lane16;

  const bf16_t* arow = A + (size_t)mrow * DMODEL;
  const bf16_t* brow = Bt + (size_t)(n0 + lane16) * DMODEL;

  v8f acc[4] = {};
  for (int ks = 0; ks < DMODEL / 32; ++ks) {
    v16bf a = load_tile16(arow + ks * 32, hi);
#pragma unroll
    for (int nt = 0; nt < 4; ++nt) {
      v16bf b = load_tile16(brow + (size_t)nt * 16 * DMODEL + ks * 32, hi);
      acc[nt] = wmma_bf16(a, b, acc[nt]);
    }
  }

#pragma unroll
  for (int nt = 0; nt < 4; ++nt) {
    const int e = n0 + nt * 16 + lane16;
    const float bv = bias[e];
#pragma unroll
    for (int r = 0; r < 8; ++r) {
      int grow = blockIdx.x * 16 + r + 8 * hi;   // global row = b*SEQ + l
      int bb = grow >> 10, l = grow & 1023;
      float val = acc[nt][r] + bv;
      if (mode == 0) {
        int h = e >> 6, d = e & 63;
        ((bf16_t*)out)[(((size_t)(bb * NHEAD + h)) * SEQ + l) * DHEAD + d] = (bf16_t)val;
      } else if (mode == 1) {
        int h = e >> 6, d = e & 63;
        ((bf16_t*)out)[(((size_t)(bb * NHEAD + h)) * DHEAD + d) * SEQ + l] = (bf16_t)val;
      } else {
        ((float*)out)[(size_t)grow * DMODEL + e] = val;
      }
    }
  }
}

// ---------------------------------------------------------------------------
// Scores: per (b,h, 16-row l-block): QK^T + relative-key skew, scaled by 1/32.
// 8 waves, 64KB LDS holds the full 16x1024 score block.
// Relative term: T = Q_blk(16x64) @ Ek[l0 .. l0+1039]^T, scatter T[r,j] into
// LDS at m = r - j + 1023 (unique addresses -> plain LDS RMW).
// Ek buffer is padded to 2048 zero-filled rows -> no bounds guard on loads.
// ---------------------------------------------------------------------------
__global__ void scores_kernel(const bf16_t* __restrict__ Q, const bf16_t* __restrict__ K,
                              const bf16_t* __restrict__ Ekb, float* __restrict__ weights) {
  const int bh = blockIdx.x >> 6;              // b*NHEAD + h
  const int l0 = (blockIdx.x & 63) << 4;
  const int t = threadIdx.x & 31, w = threadIdx.x >> 5;
  const int lane16 = t & 15, hi = t >> 4;

  __shared__ __align__(16) float s[16 * SEQ];  // 64 KB

  const bf16_t* Qh = Q + (size_t)bh * SEQ * DHEAD;
  const bf16_t* Kh = K + (size_t)bh * SEQ * DHEAD;

  // A tiles: rows l0..l0+15 of Q, k = d (contiguous 64)
  const bf16_t* qrow = Qh + (size_t)(l0 + lane16) * DHEAD;
  v16bf a0 = load_tile16(qrow, hi);            // d 0..31
  v16bf a1 = load_tile16(qrow + 32, hi);       // d 32..63

  // Phase 1: QK^T tiles -> LDS
  for (int mt = w; mt < SEQ / 16; mt += 8) {
    int m0 = mt << 4;
    const bf16_t* krow = Kh + (size_t)(m0 + lane16) * DHEAD;
    v16bf b0 = load_tile16(krow, hi);
    v16bf b1 = load_tile16(krow + 32, hi);
    v8f acc = {};
    acc = wmma_bf16(a0, b0, acc);
    acc = wmma_bf16(a1, b1, acc);
#pragma unroll
    for (int r = 0; r < 8; ++r)
      s[(r + 8 * hi) * SEQ + m0 + lane16] = acc[r];
  }
  __syncthreads();

  // Phase 2: relative-key skew.  j window [l0, l0+1039], 65 column tiles.
  for (int jt = w; jt < 65; jt += 8) {
    int jb = jt << 4;
    int jabs = l0 + jb + lane16;               // <= 2047, zero row padded
    const bf16_t* erow = Ekb + (size_t)jabs * DHEAD;
    v16bf b0 = load_tile16(erow, hi);
    v16bf b1 = load_tile16(erow + 32, hi);
    v8f acc = {};
    acc = wmma_bf16(a0, b0, acc);
    acc = wmma_bf16(a1, b1, acc);
#pragma unroll
    for (int r = 0; r < 8; ++r) {
      int rr = r + 8 * hi;                     // local l row
      int jloc = jb + lane16;
      int m = rr - jloc + (SEQ - 1);
      if (m >= 0 && m < SEQ) s[rr * SEQ + m] += acc[r];
    }
  }
  __syncthreads();

  // Phase 3: scale by 1/sqrt(EMB) = 1/32 and write to weights region
  float* outp = weights + ((size_t)bh * SEQ + l0) * SEQ;
  for (int i = threadIdx.x; i < 16 * SEQ; i += 256)
    outp[i] = s[i] * 0.03125f;
}

// ---------------------------------------------------------------------------
// Row softmax in place on weights [B*H*L rows x 1024]
// ---------------------------------------------------------------------------
__global__ void softmax_kernel(float* __restrict__ wts) {
  float* p = wts + (size_t)blockIdx.x * SEQ;
  const int t = threadIdx.x;                   // 128 threads
  __shared__ float red[128];

  float mx = -3.4e38f;
  for (int i = t; i < SEQ; i += 128) mx = fmaxf(mx, p[i]);
  red[t] = mx; __syncthreads();
  for (int sft = 64; sft > 0; sft >>= 1) {
    if (t < sft) red[t] = fmaxf(red[t], red[t + sft]);
    __syncthreads();
  }
  mx = red[0]; __syncthreads();

  float sum = 0.f;
  for (int i = t; i < SEQ; i += 128) {
    float e = __expf(p[i] - mx);
    p[i] = e;
    sum += e;
  }
  red[t] = sum; __syncthreads();
  for (int sft = 64; sft > 0; sft >>= 1) {
    if (t < sft) red[t] += red[t + sft];
    __syncthreads();
  }
  float inv = 1.0f / red[0];
  for (int i = t; i < SEQ; i += 128) p[i] *= inv;
}

// ---------------------------------------------------------------------------
// attn.V + relative-value term -> Attn bf16 [B, L, H*DH]
// Block = 4 waves, one per 16-wide d-tile; per (b,h,l-block of 16).
// The 16x1024 softmax block is staged once as bf16 in LDS (32KB); A tiles and
// the inverse-skew gather P[r,j] = W[l0+r, r-j+1023] then come from LDS.
// Main: W(16x1024) @ V(1024x64) using V^T rows (k-contig global b128 loads).
// Rel:  P(16x1056) @ Ev_window using zero-padded Ev^T rows (no guards).
// ---------------------------------------------------------------------------
__global__ void attnout_kernel(const float* __restrict__ weights, const bf16_t* __restrict__ Vt,
                               const bf16_t* __restrict__ Evt, bf16_t* __restrict__ Attn) {
  const int bh = blockIdx.x >> 6;
  const int l0 = (blockIdx.x & 63) << 4;
  const int t = threadIdx.x & 31, w = threadIdx.x >> 5;
  const int lane16 = t & 15, hi = t >> 4;
  const int d0 = w * 16;

  __shared__ __align__(16) bf16_t Wlds[16 * SEQ];   // 32 KB bf16 copy of the W block

  const float* wblk = weights + ((size_t)bh * SEQ + l0) * SEQ;
  for (int i = threadIdx.x; i < 16 * SEQ; i += 128)
    Wlds[i] = (bf16_t)wblk[i];
  __syncthreads();

  const bf16_t* wrow = Wlds + lane16 * SEQ;                              // A row (LDS)
  const bf16_t* vrow = Vt + ((size_t)bh * DHEAD + d0 + lane16) * SEQ;    // B column (V^T row)
  const bf16_t* erow = Evt + (size_t)(d0 + lane16) * EVT_STRIDE + l0;    // B column (Ev^T row)

  v8f acc = {};

  // Main term: k = m over 1024
  for (int ks = 0; ks < SEQ / 32; ++ks) {
    v16bf a = load_tile16(wrow + ks * 32, hi);
    v16bf b = load_tile16(vrow + ks * 32, hi);
    acc = wmma_bf16(a, b, acc);
  }

  // Relative-value term: jloc window [0, 1056); Evt rows zero-padded.
  for (int ks = 0; ks < 33; ++ks) {
    union { bf16_t e[16]; v16bf v; } ua;
#pragma unroll
    for (int i = 0; i < 16; ++i) {
      int koff = (i < 8) ? (hi * 8 + i) : (16 + hi * 8 + (i - 8));
      int jloc = ks * 32 + koff;
      int m = lane16 - jloc + (SEQ - 1);       // P[lane16, jloc]
      ua.e[i] = (m >= 0 && m < SEQ) ? wrow[m] : (bf16_t)0.0f;
    }
    v16bf b = load_tile16(erow + ks * 32, hi);
    acc = wmma_bf16(ua.v, b, acc);
  }

  const int b_ = bh >> 4, h = bh & 15;
#pragma unroll
  for (int r = 0; r < 8; ++r) {
    int l = l0 + r + 8 * hi;
    int e = h * DHEAD + d0 + lane16;
    Attn[((size_t)(b_ * SEQ + l)) * DMODEL + e] = (bf16_t)acc[r];
  }
}

// ---------------------------------------------------------------------------
// Host launcher
// ---------------------------------------------------------------------------
extern "C" void kernel_launch(void* const* d_in, const int* in_sizes, int n_in,
                              void* d_out, int out_size, void* d_ws, size_t ws_size,
                              hipStream_t stream) {
  const float* queries = (const float*)d_in[0];
  const float* keys    = (const float*)d_in[1];
  const float* values  = (const float*)d_in[2];
  const float* Wq = (const float*)d_in[3];  const float* bq = (const float*)d_in[4];
  const float* Wk = (const float*)d_in[5];  const float* bk = (const float*)d_in[6];
  const float* Wv = (const float*)d_in[7];  const float* bv = (const float*)d_in[8];
  const float* Wo = (const float*)d_in[9];  const float* bo = (const float*)d_in[10];
  const float* Ek = (const float*)d_in[11];
  const float* Ev = (const float*)d_in[12];

  float* out_f32 = (float*)d_out;                               // [B, L, D]
  float* weights = out_f32 + (size_t)BATCH * SEQ * DMODEL;      // [B, H, L, L]

  // Workspace layout (bf16 buffers), 256B aligned
  char* base = (char*)d_ws;
  size_t off = 0;
  auto take = [&](size_t bytes) {
    char* p = base + off;
    off += (bytes + 255) & ~(size_t)255;
    return p;
  };
  const size_t NTOK = (size_t)BATCH * SEQ;                      // 4096
  bf16_t* Xq  = (bf16_t*)take(NTOK * DMODEL * 2);
  bf16_t* Xk  = (bf16_t*)take(NTOK * DMODEL * 2);
  bf16_t* Xv  = (bf16_t*)take(NTOK * DMODEL * 2);
  bf16_t* Wqt = (bf16_t*)take((size_t)DMODEL * DMODEL * 2);
  bf16_t* Wkt = (bf16_t*)take((size_t)DMODEL * DMODEL * 2);
  bf16_t* Wvt = (bf16_t*)take((size_t)DMODEL * DMODEL * 2);
  bf16_t* Wot = (bf16_t*)take((size_t)DMODEL * DMODEL * 2);
  bf16_t* Ekb = (bf16_t*)take((size_t)EKROWS * DHEAD * 2);      // padded, zero row 2047
  bf16_t* Evt = (bf16_t*)take((size_t)DHEAD * EVT_STRIDE * 2);  // padded rows
  bf16_t* Qb  = (bf16_t*)take(NTOK * DMODEL * 2);               // [B,H,L,DH]
  bf16_t* Kb  = (bf16_t*)take(NTOK * DMODEL * 2);
  bf16_t* Vtb = (bf16_t*)take(NTOK * DMODEL * 2);               // [B,H,DH,L]
  bf16_t* Att = (bf16_t*)take(NTOK * DMODEL * 2);               // [B,L,D]
  (void)ws_size; (void)in_sizes; (void)n_in; (void)out_size;

  // Zero-fill padded embedding buffers first, then overwrite valid entries.
  const int nEk = EKROWS * DHEAD;                               // 131072
  const int nEv = DHEAD * EVT_STRIDE;                           // 135168
  fill0_bf16_kernel<<<(nEk + 255) / 256, 256, 0, stream>>>(Ekb, nEk);
  fill0_bf16_kernel<<<(nEv + 255) / 256, 256, 0, stream>>>(Evt, nEv);

  const int nTok = (int)(NTOK * DMODEL);                        // 4194304
  cvt_kernel<<<(nTok + 255) / 256, 256, 0, stream>>>(queries, Xq, nTok);
  cvt_kernel<<<(nTok + 255) / 256, 256, 0, stream>>>(keys,    Xk, nTok);
  cvt_kernel<<<(nTok + 255) / 256, 256, 0, stream>>>(values,  Xv, nTok);
  cvt_kernel<<<(NREL * DHEAD + 255) / 256, 256, 0, stream>>>(Ek, Ekb, NREL * DHEAD);

  const int nW = DMODEL * DMODEL;
  transpose_cvt_kernel<<<(nW + 255) / 256, 256, 0, stream>>>(Wq, Wqt, DMODEL, DMODEL, DMODEL);
  transpose_cvt_kernel<<<(nW + 255) / 256, 256, 0, stream>>>(Wk, Wkt, DMODEL, DMODEL, DMODEL);
  transpose_cvt_kernel<<<(nW + 255) / 256, 256, 0, stream>>>(Wv, Wvt, DMODEL, DMODEL, DMODEL);
  transpose_cvt_kernel<<<(nW + 255) / 256, 256, 0, stream>>>(Wo, Wot, DMODEL, DMODEL, DMODEL);
  transpose_cvt_kernel<<<(NREL * DHEAD + 255) / 256, 256, 0, stream>>>(Ev, Evt, NREL, DHEAD,
                                                                       EVT_STRIDE);

  dim3 ggrid(NTOK / 16, DMODEL / 256);                          // (256, 4)
  gemm_bf16_kernel<<<ggrid, 128, 0, stream>>>(Xq, Wqt, bq, Qb, 0);
  gemm_bf16_kernel<<<ggrid, 128, 0, stream>>>(Xk, Wkt, bk, Kb, 0);
  gemm_bf16_kernel<<<ggrid, 128, 0, stream>>>(Xv, Wvt, bv, Vtb, 1);

  scores_kernel<<<BATCH * NHEAD * (SEQ / 16), 256, 0, stream>>>(Qb, Kb, Ekb, weights);
  softmax_kernel<<<BATCH * NHEAD * SEQ, 128, 0, stream>>>(weights);
  attnout_kernel<<<BATCH * NHEAD * (SEQ / 16), 128, 0, stream>>>(weights, Vtb, Evt, Att);

  gemm_bf16_kernel<<<ggrid, 128, 0, stream>>>(Att, Wot, bo, out_f32, 2);
}
